// GraphEncoderLayer_31791347925244
// MI455X (gfx1250) — matrix operand order repaired and checked by reading, not compiled, not run on hardware
//
#include <hip/hip_runtime.h>
#include <hip/hip_bf16.h>

typedef _Float16 f16;
typedef __attribute__((ext_vector_type(4)))  _Float16 v4h;
typedef __attribute__((ext_vector_type(8)))  _Float16 v8h;
typedef __attribute__((ext_vector_type(16))) _Float16 v16h;
typedef __attribute__((ext_vector_type(8)))  float    v8f;

#define CAT16(lo, hi) __builtin_shufflevector(lo, hi, 0,1,2,3,4,5,6,7,8,9,10,11,12,13,14,15)

__device__ __forceinline__ float lrelu(float x) { return x > 0.f ? x : 0.2f * x; }
__device__ __forceinline__ float sigm(float x) { return 1.0f / (1.0f + __expf(-x)); }

// ---------------------------------------------------------------------------
// Weight prep: pack f32 weights into f16 N-major (transposed) layouts for WMMA
// ---------------------------------------------------------------------------
__global__ void prep_wio(const float* __restrict__ W_in, const float* __restrict__ W_out,
                         f16* __restrict__ WioT) {
  int idx = blockIdx.x * 256 + threadIdx.x;
  if (idx >= 1024 * 1536) return;
  int n = idx / 1536, k = idx % 1536;
  float v = (n < 512) ? W_in[k * 512 + n] : W_out[k * 512 + (n - 512)];
  WioT[idx] = (f16)v;
}

__global__ void prep_wrzt(const float* __restrict__ W_r, const float* __restrict__ W_z,
                          const float* __restrict__ W_t, f16* __restrict__ WrztT) {
  int idx = blockIdx.x * 256 + threadIdx.x;
  if (idx >= 1536 * 1536) return;
  int n = idx / 1536, k = idx % 1536;
  float v;
  if (n < 512)       v = W_r[k * 512 + n];
  else if (n < 1024) v = W_z[k * 512 + (n - 512)];
  else               v = (k < 1024) ? W_t[k * 512 + (n - 1024)] : 0.0f;
  WrztT[idx] = (f16)v;
}

__global__ void prep_wt2(const float* __restrict__ W_t, f16* __restrict__ Wt2T) {
  int idx = blockIdx.x * 256 + threadIdx.x;
  if (idx >= 512 * 512) return;
  int n = idx / 512, k = idx % 512;
  Wt2T[idx] = (f16)W_t[(1024 + k) * 512 + n];
}

__global__ void prep_wv(const float* __restrict__ W_in, const float* __restrict__ W_out,
                        const float* __restrict__ a_in_v, const float* __restrict__ a_out_v,
                        float* __restrict__ wv_in, float* __restrict__ wv_out) {
  __shared__ float red0[128], red1[128];
  int k = blockIdx.x, tid = threadIdx.x;
  float s0 = 0.f, s1 = 0.f;
  for (int e = tid; e < 512; e += 128) {
    s0 += W_in[k * 512 + e] * a_in_v[e];
    s1 += W_out[k * 512 + e] * a_out_v[e];
  }
  red0[tid] = s0; red1[tid] = s1;
  __syncthreads();
  for (int off = 64; off > 0; off >>= 1) {
    if (tid < off) { red0[tid] += red0[tid + off]; red1[tid] += red1[tid + off]; }
    __syncthreads();
  }
  if (tid == 0) { wv_in[k] = red0[0]; wv_out[k] = red1[0]; }
}

// ---------------------------------------------------------------------------
// Node prep: masked Xm (8192 x 1536 f16) and nodes block of Abig2
// ---------------------------------------------------------------------------
__global__ void prep_nodes(const float* __restrict__ nodes, const int* __restrict__ node_type,
                           f16* __restrict__ Xm, f16* __restrict__ Abig2) {
  int idx = blockIdx.x * 256 + threadIdx.x;
  if (idx >= 8192 * 512) return;
  int row = idx >> 9, e = idx & 511;
  int t = node_type[row] - 2;
  f16 h = (f16)nodes[idx];
  Abig2[(size_t)row * 1536 + 1024 + e] = h;
  size_t base = (size_t)row * 1536 + e;
  Xm[base]        = (t == 0) ? h : (f16)0.f;
  Xm[base + 512]  = (t == 1) ? h : (f16)0.f;
  Xm[base + 1024] = (t == 2) ? h : (f16)0.f;
}

__global__ void row_dots(const float* __restrict__ nodes, const f16* __restrict__ Xm,
                         const float* __restrict__ a_in_q, const float* __restrict__ a_out_q,
                         const float* __restrict__ wv_in, const float* __restrict__ wv_out,
                         float* __restrict__ sq_in, float* __restrict__ sq_out,
                         float* __restrict__ sv_in, float* __restrict__ sv_out) {
  __shared__ float red[4][128];
  int row = blockIdx.x, tid = threadIdx.x;
  float s0 = 0.f, s1 = 0.f, s2 = 0.f, s3 = 0.f;
  const float* nr = nodes + (size_t)row * 512;
  for (int e = tid; e < 512; e += 128) {
    float x = nr[e];
    s0 += x * a_in_q[e];
    s1 += x * a_out_q[e];
  }
  const f16* xr = Xm + (size_t)row * 1536;
  for (int k = tid; k < 1536; k += 128) {
    float x = (float)xr[k];
    s2 += x * wv_in[k];
    s3 += x * wv_out[k];
  }
  red[0][tid] = s0; red[1][tid] = s1; red[2][tid] = s2; red[3][tid] = s3;
  __syncthreads();
  for (int off = 64; off > 0; off >>= 1) {
    if (tid < off)
      for (int q = 0; q < 4; ++q) red[q][tid] += red[q][tid + off];
    __syncthreads();
  }
  if (tid == 0) {
    sq_in[row] = red[0][0]; sq_out[row] = red[1][0];
    sv_in[row] = red[2][0]; sv_out[row] = red[3][0];
  }
}

// ---------------------------------------------------------------------------
// WMMA GEMM with register double-buffering of fragments.
// Block: 256 thr = 8 waves; block tile 64M x 128N; wave tile 16M x 64N.
// MODE 0: f32 store to C (ldc). MODE 1: f16 transposed store to CT (ldct).
// ---------------------------------------------------------------------------
template <int MODE>
__global__ __launch_bounds__(256) void wmma_gemm(
    const f16* __restrict__ A, const f16* __restrict__ BT,
    float* __restrict__ C, f16* __restrict__ CT,
    int K, int lda, int ldb, int ldc, int ldct) {
  int tid = threadIdx.x;
  int lane = tid & 31;
  int wave = tid >> 5;
  int rowBase = blockIdx.x * 64 + (wave >> 1) * 16;
  int nBase0  = blockIdx.y * 128 + (wave & 1) * 64;
  int ar = lane & 15;
  int ak = (lane >> 4) << 3;   // 0 or 8
  int bn = lane & 15;
  int bk = (lane >> 4) << 3;   // 0 or 8

  const f16* Arow = A + (size_t)(rowBase + ar) * lda + ak;
  const f16* Bp[4];
#pragma unroll
  for (int f = 0; f < 4; ++f)
    Bp[f] = BT + (size_t)(nBase0 + f * 16 + bn) * ldb + bk;

  v8f acc[4] = {};
  v16h a_cur, b_cur[4];
  {
    v8h lo = *(const v8h*)(Arow);
    v8h hi = *(const v8h*)(Arow + 16);
    a_cur = CAT16(lo, hi);
#pragma unroll
    for (int f = 0; f < 4; ++f) {
      v8h blo = *(const v8h*)(Bp[f]);
      v8h bhi = *(const v8h*)(Bp[f] + 16);
      b_cur[f] = CAT16(blo, bhi);
    }
  }
  for (int k = 32; k <= K; k += 32) {
    v16h a_nxt = {};
    v16h b_nxt[4] = {};
    if (k < K) {  // issue next-step loads before consuming current fragments
      v8h lo = *(const v8h*)(Arow + k);
      v8h hi = *(const v8h*)(Arow + k + 16);
      a_nxt = CAT16(lo, hi);
#pragma unroll
      for (int f = 0; f < 4; ++f) {
        v8h blo = *(const v8h*)(Bp[f] + k);
        v8h bhi = *(const v8h*)(Bp[f] + k + 16);
        b_nxt[f] = CAT16(blo, bhi);
      }
    }
#pragma unroll
    for (int f = 0; f < 4; ++f)
      acc[f] = __builtin_amdgcn_wmma_f32_16x16x32_f16(
          false, a_cur, false, b_cur[f], (short)0, acc[f], false, false);
    a_cur = a_nxt;
#pragma unroll
    for (int f = 0; f < 4; ++f) b_cur[f] = b_nxt[f];
  }

  int cn = lane & 15;
  int cm = (lane >> 4) << 3;
#pragma unroll
  for (int f = 0; f < 4; ++f)
#pragma unroll
    for (int r = 0; r < 8; ++r) {
      float v = acc[f][r];
      int row = rowBase + cm + r;
      int col = nBase0 + f * 16 + cn;
      if (MODE == 0) C[(size_t)row * ldc + col] = v;
      else           CT[(size_t)col * ldct + row] = (f16)v;
    }
}

// ---------------------------------------------------------------------------
// GAT attention: block = 128 thr (4 waves) per (dir, batch, 16-row) tile.
// sv staged to LDS via async global->LDS copy (ASYNCcnt path). Per jb-step:
// clause of B-fragment loads, then a straight-line 4-wide P-tile build
// (int4 mask load, branchless selects, v4h LDS store) hides their latency,
// then 8 back-to-back WMMAs.
// ---------------------------------------------------------------------------
__global__ __launch_bounds__(128) void gat_attn(
    const int* __restrict__ mask,
    const float* __restrict__ sq_in, const float* __restrict__ sq_out,
    const float* __restrict__ sv_in, const float* __restrict__ sv_out,
    const f16* __restrict__ VT, f16* __restrict__ Abig2) {
  int dir = blockIdx.y;
  int b = blockIdx.x >> 6;
  int i0 = (blockIdx.x & 63) << 4;
  const float* sq = dir ? sq_out : sq_in;
  const float* sv = dir ? sv_out : sv_in;
  const int* msk = mask + (((size_t)dir * 8 + b) * 1024 + i0) * 1024;
  const f16* Vb = VT + (size_t)(dir ? 512 : 0) * 8192 + (size_t)b * 1024;

  __shared__ __align__(16) float sv_s[1024];
  __shared__ float sq_s[16];
  __shared__ float Sinv[16];
  __shared__ float red[128];
  __shared__ __align__(16) f16 P_s[16][32];

  int tid = threadIdx.x;
  // async global -> LDS copy of sv row block (4 KB): 128 lanes x 16B x 2
  {
    unsigned lds0 = (unsigned)(size_t)(&sv_s[0]) + (unsigned)(tid * 16);
    unsigned long long g0 = (unsigned long long)(const void*)(sv + (size_t)b * 1024 + tid * 4);
    asm volatile("global_load_async_to_lds_b128 %0, %1, off"
                 :: "v"(lds0), "v"(g0) : "memory");
    asm volatile("global_load_async_to_lds_b128 %0, %1, off offset:2048"
                 :: "v"(lds0), "v"(g0) : "memory");
  }
  if (tid < 16) sq_s[tid] = sq[b * 1024 + i0 + tid];
  asm volatile("s_wait_asynccnt 0x0" ::: "memory");
  __syncthreads();

  { // row sums of masked exp: thread owns row (tid&15), 4-wide j chunks
    int r = tid & 15, jo = (tid >> 4) << 2;           // 8 chunk-lanes per row
    const int* mrow = msk + (size_t)r * 1024;
    float q = sq_s[r], s = 0.f;
    for (int j0 = jo; j0 < 1024; j0 += 32) {
      int4 m = *(const int4*)&mrow[j0];
      float e0 = __expf(lrelu(q + sv_s[j0 + 0]));
      float e1 = __expf(lrelu(q + sv_s[j0 + 1]));
      float e2 = __expf(lrelu(q + sv_s[j0 + 2]));
      float e3 = __expf(lrelu(q + sv_s[j0 + 3]));
      s += (m.x > 0 ? e0 : 0.f) + (m.y > 0 ? e1 : 0.f) +
           (m.z > 0 ? e2 : 0.f) + (m.w > 0 ? e3 : 0.f);
    }
    red[tid] = s;
    __syncthreads();
    if (tid < 16) {
      float t = 0.f;
      for (int k = 0; k < 8; ++k) t += red[tid + (k << 4)];
      Sinv[tid] = 1.0f / fmaxf(t, 1e-30f);
    }
    __syncthreads();
  }

  int lane = tid & 31, wave = tid >> 5;
  int ar = lane & 15, ak = (lane >> 4) << 3;
  int bn = lane & 15, bko = (lane >> 4) << 3;
  const f16* Bbase = Vb + (size_t)(wave * 128 + bn) * 8192 + bko;
  int pr = tid >> 3;                // row 0..15
  int pc = (tid & 7) << 2;          // col 0,4,...,28
  const int* prow = msk + (size_t)pr * 1024 + pc;
  float pq = sq_s[pr];
  float psi = Sinv[pr];
  v8f acc[8] = {};

  for (int jb = 0; jb < 1024; jb += 32) {
    // issue all B-fragment loads for this K-chunk first (independent of P_s)
    v16h bfrag[8];
#pragma unroll
    for (int f = 0; f < 8; ++f) {
      const f16* Bp = Bbase + (size_t)(f * 16) * 8192 + jb;
      v8h blo = *(const v8h*)Bp;
      v8h bhi = *(const v8h*)(Bp + 16);
      bfrag[f] = CAT16(blo, bhi);
    }
    // straight-line 4-wide probability tile build while loads are in flight
    {
      int j = jb + pc;
      int4 m = *(const int4*)prow;
      prow += 32;
      float e0 = __expf(lrelu(pq + sv_s[j + 0])) * psi;
      float e1 = __expf(lrelu(pq + sv_s[j + 1])) * psi;
      float e2 = __expf(lrelu(pq + sv_s[j + 2])) * psi;
      float e3 = __expf(lrelu(pq + sv_s[j + 3])) * psi;
      v4h pv;
      pv[0] = (f16)(m.x > 0 ? e0 : 0.f);
      pv[1] = (f16)(m.y > 0 ? e1 : 0.f);
      pv[2] = (f16)(m.z > 0 ? e2 : 0.f);
      pv[3] = (f16)(m.w > 0 ? e3 : 0.f);
      *(v4h*)&P_s[pr][pc] = pv;
    }
    __syncthreads();
    v8h alo = *(const v8h*)&P_s[ar][ak];
    v8h ahi = *(const v8h*)&P_s[ar][ak + 16];
    v16h afrag = CAT16(alo, ahi);
#pragma unroll
    for (int f = 0; f < 8; ++f)
      acc[f] = __builtin_amdgcn_wmma_f32_16x16x32_f16(
          false, afrag, false, bfrag[f], (short)0, acc[f], false, false);
    __syncthreads();
  }

  int cn = lane & 15, cm = (lane >> 4) << 3;
  int colBase = (dir ? 512 : 0) + wave * 128;
#pragma unroll
  for (int f = 0; f < 8; ++f)
#pragma unroll
    for (int r = 0; r < 8; ++r)
      Abig2[(size_t)(b * 1024 + i0 + cm + r) * 1536 + colBase + f * 16 + cn] =
          (f16)acc[f][r];
}

// ---------------------------------------------------------------------------
// r = sigmoid(P[:, :512] + b_r); rn = f16(r * nodes)   (float4 vectorized)
// ---------------------------------------------------------------------------
__global__ void gate_r(const float* __restrict__ P, const float* __restrict__ b_r,
                       const float* __restrict__ nodes, f16* __restrict__ rn) {
  int idx4 = blockIdx.x * 256 + threadIdx.x;
  if (idx4 >= 8192 * 128) return;
  int row = idx4 >> 7, e4 = (idx4 & 127) << 2;
  float4 p = *(const float4*)&P[(size_t)row * 1536 + e4];
  float4 bb = *(const float4*)&b_r[e4];
  float4 x = *(const float4*)&nodes[(size_t)row * 512 + e4];
  v4h o;
  o[0] = (f16)(sigm(p.x + bb.x) * x.x);
  o[1] = (f16)(sigm(p.y + bb.y) * x.y);
  o[2] = (f16)(sigm(p.z + bb.z) * x.z);
  o[3] = (f16)(sigm(p.w + bb.w) * x.w);
  *(v4h*)&rn[(size_t)row * 512 + e4] = o;
}

// ---------------------------------------------------------------------------
// out = (1-z)*nodes + z*tanh(Pq + H2 + b_t), z = sigmoid(Pz + b_z)  (float4)
// ---------------------------------------------------------------------------
__global__ void gate_out(const float* __restrict__ P, const float* __restrict__ H2,
                         const float* __restrict__ b_z, const float* __restrict__ b_t,
                         const float* __restrict__ nodes, float* __restrict__ out) {
  int idx4 = blockIdx.x * 256 + threadIdx.x;
  if (idx4 >= 8192 * 128) return;
  int row = idx4 >> 7, e4 = (idx4 & 127) << 2;
  float4 pz = *(const float4*)&P[(size_t)row * 1536 + 512 + e4];
  float4 pq = *(const float4*)&P[(size_t)row * 1536 + 1024 + e4];
  float4 h2 = *(const float4*)&H2[(size_t)row * 512 + e4];
  float4 bz = *(const float4*)&b_z[e4];
  float4 bt = *(const float4*)&b_t[e4];
  float4 x = *(const float4*)&nodes[(size_t)row * 512 + e4];
  float4 o;
  {
    float z = sigm(pz.x + bz.x); float hh = tanhf(pq.x + h2.x + bt.x);
    o.x = (1.0f - z) * x.x + z * hh;
  }
  {
    float z = sigm(pz.y + bz.y); float hh = tanhf(pq.y + h2.y + bt.y);
    o.y = (1.0f - z) * x.y + z * hh;
  }
  {
    float z = sigm(pz.z + bz.z); float hh = tanhf(pq.z + h2.z + bt.z);
    o.z = (1.0f - z) * x.z + z * hh;
  }
  {
    float z = sigm(pz.w + bz.w); float hh = tanhf(pq.w + h2.w + bt.w);
    o.w = (1.0f - z) * x.w + z * hh;
  }
  *(float4*)&out[(size_t)row * 512 + e4] = o;
}

// ---------------------------------------------------------------------------
extern "C" void kernel_launch(void* const* d_in, const int* in_sizes, int n_in,
                              void* d_out, int out_size, void* d_ws, size_t ws_size,
                              hipStream_t stream) {
  const float* nodes   = (const float*)d_in[0];
  const int*   mask    = (const int*)d_in[1];
  const int*   ntype   = (const int*)d_in[2];
  const float* W_in    = (const float*)d_in[3];
  const float* W_out   = (const float*)d_in[4];
  const float* a_in_q  = (const float*)d_in[5];
  const float* a_in_v  = (const float*)d_in[6];
  const float* a_out_q = (const float*)d_in[7];
  const float* a_out_v = (const float*)d_in[8];
  const float* W_r     = (const float*)d_in[9];
  const float* b_r     = (const float*)d_in[10];
  const float* W_z     = (const float*)d_in[11];
  const float* b_z     = (const float*)d_in[12];
  const float* W_t     = (const float*)d_in[13];
  const float* b_t     = (const float*)d_in[14];
  float* out = (float*)d_out;

  char* ws = (char*)d_ws;
  const size_t R = 8192;
  f16*   Xm     = (f16*)(ws + 0);                    // R*1536 f16
  f16*   WioT   = (f16*)(ws + 25165824);             // 1024*1536 f16
  f16*   WrztT  = (f16*)(ws + 28311552);             // 1536*1536 f16
  f16*   Wt2T   = (f16*)(ws + 33030144);             // 512*512 f16
  f16*   VT     = (f16*)(ws + 33554432);             // 1024*R f16
  float* sq_in  = (float*)(ws + 50331648);
  float* sq_out = (float*)(ws + 50364416);
  float* sv_in  = (float*)(ws + 50397184);
  float* sv_out = (float*)(ws + 50429952);
  float* wv_in  = (float*)(ws + 50462720);
  float* wv_out = (float*)(ws + 50468864);
  f16*   Abig2  = (f16*)(ws + 50475008);             // R*1536 f16
  float* P      = (float*)(ws + 75640832);           // R*1536 f32
  f16*   rn     = (f16*)(ws + 125972480);            // R*512 f16
  float* H2     = (float*)(ws + 134361088);          // R*512 f32
  (void)ws_size; (void)n_in; (void)in_sizes; (void)out_size;

  prep_wio <<<(1024 * 1536 + 255) / 256, 256, 0, stream>>>(W_in, W_out, WioT);
  prep_wrzt<<<(1536 * 1536 + 255) / 256, 256, 0, stream>>>(W_r, W_z, W_t, WrztT);
  prep_wt2 <<<(512 * 512 + 255) / 256, 256, 0, stream>>>(W_t, Wt2T);
  prep_wv  <<<1536, 128, 0, stream>>>(W_in, W_out, a_in_v, a_out_v, wv_in, wv_out);

  prep_nodes<<<(8192 * 512 + 255) / 256, 256, 0, stream>>>(nodes, ntype, Xm, Abig2);

  row_dots<<<8192, 128, 0, stream>>>(nodes, Xm, a_in_q, a_out_q, wv_in, wv_out,
                                     sq_in, sq_out, sv_in, sv_out);

  // GEMM1: V_T = (Xm @ W_io)^T  (8192 x 1024 x 1536), f16 transposed store
  wmma_gemm<1><<<dim3(128, 8), 256, 0, stream>>>(Xm, WioT, nullptr, VT,
                                                 1536, 1536, 1536, 0, (int)R);

  // attention (dir 0 = in, 1 = out) -> h_in/h_out f16 into Abig2
  gat_attn<<<dim3(512, 2), 128, 0, stream>>>(mask, sq_in, sq_out, sv_in, sv_out,
                                             VT, Abig2);

  // GEMM2: P = Abig2 @ [W_r|W_z|W_t(top)]  (8192 x 1536 x 1536), f32
  wmma_gemm<0><<<dim3(128, 12), 256, 0, stream>>>(Abig2, WrztT, P, nullptr,
                                                  1536, 1536, 1536, 1536, 0);

  gate_r<<<(8192 * 128 + 255) / 256, 256, 0, stream>>>(P, b_r, nodes, rn);

  // GEMM3: H2 = rn @ W_t(bottom)  (8192 x 512 x 512), f32
  wmma_gemm<0><<<dim3(128, 4), 256, 0, stream>>>(rn, Wt2T, H2, nullptr,
                                                 512, 512, 512, 512, 0);

  gate_out<<<(8192 * 128 + 255) / 256, 256, 0, stream>>>(P, H2, b_z, b_t, nodes, out);
}
